// LinearSelfAttentionBlock_75076028334557
// MI455X (gfx1250) — matrix operand order repaired and checked by reading, not compiled, not run on hardware
//
#include <hip/hip_runtime.h>

// Problem constants from the reference.
#define BB 4
#define LL 4096
#define EE 1024
#define HH 16
#define DD 64
// M_total = BB*LL = 16384, sqrt(D) = 8

typedef __bf16 bf16_t;
typedef bf16_t v16bf __attribute__((ext_vector_type(16)));
typedef bf16_t v8bf  __attribute__((ext_vector_type(8)));
typedef float  v8f   __attribute__((ext_vector_type(8)));

struct Frag32 { v8bf lo, hi; };  // 32 bytes, 16-byte aligned halves

__device__ __forceinline__ float gfun(float x) {
  // (softplus(5x)/5)^2, numerically stable
  float y = 5.0f * x;
  float sp = (y > 20.0f) ? y : __logf(1.0f + __expf(y));
  sp *= 0.2f;
  return sp * sp;
}

__device__ __forceinline__ v8f wmma_bf16(v16bf a, v16bf b, v8f c) {
  return __builtin_amdgcn_wmma_f32_16x16x32_bf16(false, a, false, b,
                                                 (short)0, c, false, false);
}

__device__ __forceinline__ void zero_acc(v8f& a) {
#pragma unroll
  for (int r = 0; r < 8; ++r) a[r] = 0.0f;
}

// Load one WMMA operand fragment: 16 contiguous bf16 (two ds_load_b128).
__device__ __forceinline__ v16bf load_frag(const bf16_t* p) {
  return __builtin_bit_cast(v16bf, *reinterpret_cast<const Frag32*>(p));
}

// Pack two floats into adjacent bf16 and store as one dword to LDS.
__device__ __forceinline__ void store2(bf16_t* p, float a, float b) {
  union { bf16_t h[2]; unsigned u; } t;
  t.h[0] = (bf16_t)a; t.h[1] = (bf16_t)b;
  *reinterpret_cast<unsigned*>(p) = t.u;
}

// Fragment-major layouts. A-type (16x32 MxK operand), per tile of 16 rows:
//   lane = (m&15) + 16*((k>>3)&1),  elem e = (k&7) | ((k>>4)<<3)
// B-type (32x16 KxN operand), per tile of 16 cols:
//   lane = (n&15) + 16*((k>>3)&1),  elem e = (k&7) | ((k>>4)<<3)
// Storage: lds[(frag*32 + lane)*STRIDE + e], e contiguous -> b128 reads.

// ---------------------------------------------------------------------------
// Kernel 1: QKV projection GEMM. out = x @ W + b, W selected by blockIdx.z.
// 256 threads / 8 waves; block tile 128(M) x 64(N); wave tile 32x32.
// ---------------------------------------------------------------------------
#define ASTRIDE 24  // 48B per lane, 16B aligned, low bank conflict
__global__ __launch_bounds__(256) void qkv_gemm(
    const float* __restrict__ x,
    const float* __restrict__ Wq, const float* __restrict__ bq,
    const float* __restrict__ Wk, const float* __restrict__ bk,
    const float* __restrict__ Wv, const float* __restrict__ bv,
    float* __restrict__ Q, float* __restrict__ K, float* __restrict__ V)
{
  __shared__ bf16_t As2[8 * 32 * ASTRIDE];  // 8 row-tiles  (12 KB)
  __shared__ bf16_t Bs2[4 * 32 * ASTRIDE];  // 4 col-tiles  ( 6 KB)

  const int which = blockIdx.z;
  const float* W    = (which == 0) ? Wq : ((which == 1) ? Wk : Wv);
  const float* bias = (which == 0) ? bq : ((which == 1) ? bk : bv);
  float* outp       = (which == 0) ? Q  : ((which == 1) ? K  : V);

  const int m0 = blockIdx.y * 128;
  const int n0 = blockIdx.x * 64;
  const int tid = threadIdx.x;
  const int lane = tid & 31;
  const int w = tid >> 5;
  const int wm = w & 3;   // wave M position (4)
  const int wn = w >> 2;  // wave N position (2)
  const int mhalf = lane >> 4;
  const int l15 = lane & 15;

  v8f acc[2][2];
#pragma unroll
  for (int i = 0; i < 2; ++i)
#pragma unroll
    for (int j = 0; j < 2; ++j) zero_acc(acc[i][j]);

  for (int k0 = 0; k0 < EE; k0 += 32) {
    if (k0 + 32 < EE) {
      __builtin_prefetch(&x[(m0 + (tid >> 1)) * EE + k0 + 32], 0, 1);
      __builtin_prefetch(&W[(k0 + 32 + (tid & 31)) * EE + n0], 0, 1);
    }
    // Stage A: x[m0+row][k0+c..c+1] as float2, write packed b32 in frag order
#pragma unroll
    for (int j = 0; j < 8; ++j) {
      int idx = j * 256 + tid;
      int row = idx >> 4, c = (idx & 15) * 2;
      float2 xv = *reinterpret_cast<const float2*>(&x[(m0 + row) * EE + k0 + c]);
      int lane_s = (row & 15) + (((c >> 3) & 1) << 4);
      int e = (c & 7) | ((c >> 4) << 3);
      int frag = row >> 4;
      store2(&As2[(frag * 32 + lane_s) * ASTRIDE + e], xv.x, xv.y);
    }
    // Stage B: W[k0+k..k+1][n0+n], coalesced over n, packed along k
#pragma unroll
    for (int j = 0; j < 4; ++j) {
      int idx = j * 256 + tid;
      int n = idx & 63, k = (idx >> 6) * 2;
      float w0 = W[(k0 + k) * EE + n0 + n];
      float w1 = W[(k0 + k + 1) * EE + n0 + n];
      int lane_s = (n & 15) + (((k >> 3) & 1) << 4);
      int frag = n >> 4;
      store2(&Bs2[(frag * 32 + lane_s) * ASTRIDE + ((k & 7) | ((k >> 4) << 3))], w0, w1);
    }
    __syncthreads();

#pragma unroll
    for (int ti = 0; ti < 2; ++ti) {
      v16bf afrag = load_frag(&As2[((wm * 2 + ti) * 32 + lane) * ASTRIDE]);
#pragma unroll
      for (int tj = 0; tj < 2; ++tj) {
        v16bf bfrag = load_frag(&Bs2[((wn * 2 + tj) * 32 + lane) * ASTRIDE]);
        acc[ti][tj] = wmma_bf16(afrag, bfrag, acc[ti][tj]);
      }
    }
    __syncthreads();
  }

  // C layout: VGPR r -> M = r + 8*(lane>>4); N = lane&15
#pragma unroll
  for (int ti = 0; ti < 2; ++ti)
#pragma unroll
    for (int tj = 0; tj < 2; ++tj) {
      int n = n0 + wn * 32 + tj * 16 + l15;
      float bsc = bias[n];
#pragma unroll
      for (int r = 0; r < 8; ++r) {
        int m = m0 + wm * 32 + ti * 16 + r + 8 * mhalf;
        outp[m * EE + n] = acc[ti][tj][r] + bsc;
      }
    }
}

// ---------------------------------------------------------------------------
// Kernel 2: q_global[b,h,d] = mean_l Q[b,l,h,d] / sqrt(D). One block per (b,h).
// ---------------------------------------------------------------------------
__global__ __launch_bounds__(256) void qglobal_kernel(
    const float* __restrict__ Q, float* __restrict__ qg)
{
  __shared__ float red[4][64];
  int bh = blockIdx.x;
  int b = bh >> 4, h = bh & 15;
  int d = threadIdx.x & 63;
  int part = threadIdx.x >> 6;
  float s = 0.f;
  int base = (b * LL) * EE + h * DD + d;
  for (int l = part * 1024; l < (part + 1) * 1024; ++l) s += Q[base + l * EE];
  red[part][d] = s;
  __syncthreads();
  if (part == 0) {
    float t = red[0][d] + red[1][d] + red[2][d] + red[3][d];
    qg[bh * DD + d] = t / (4096.0f * 8.0f);  // /L (mean) then /sqrt(D)
  }
}

// ---------------------------------------------------------------------------
// Kernel 3: logits = qg . K, softmax over L, alpha = softmax*L, K *= alpha.
// One block per (b,h); full logits row in LDS (16 KB).
// ---------------------------------------------------------------------------
__global__ __launch_bounds__(256) void softmax_scale_k(
    const float* __restrict__ qg, float* __restrict__ K)
{
  __shared__ float logits[LL];
  __shared__ float red[256];
  __shared__ float qgl[DD];
  int bh = blockIdx.x;
  int b = bh >> 4, h = bh & 15;
  int tid = threadIdx.x;
  if (tid < DD) qgl[tid] = qg[bh * DD + tid];
  __syncthreads();

  float lmax = -3.4e38f;
  for (int l = tid; l < LL; l += 256) {
    const float* kp = K + ((b * LL + l) * EE + h * DD);
    float dot = 0.f;
#pragma unroll
    for (int d = 0; d < DD; ++d) dot += qgl[d] * kp[d];
    logits[l] = dot;
    lmax = fmaxf(lmax, dot);
  }
  red[tid] = lmax;
  __syncthreads();
  for (int s = 128; s > 0; s >>= 1) {
    if (tid < s) red[tid] = fmaxf(red[tid], red[tid + s]);
    __syncthreads();
  }
  float gmax = red[0];
  __syncthreads();
  float lsum = 0.f;
  for (int l = tid; l < LL; l += 256) {
    float e = __expf(logits[l] - gmax);
    logits[l] = e;
    lsum += e;
  }
  red[tid] = lsum;
  __syncthreads();
  for (int s = 128; s > 0; s >>= 1) {
    if (tid < s) red[tid] += red[tid + s];
    __syncthreads();
  }
  float factor = (float)LL / red[0];
  __syncthreads();
  for (int idx = tid; idx < LL * DD; idx += 256) {
    int l = idx >> 6, d = idx & 63;
    K[(b * LL + l) * EE + h * DD + d] *= logits[l] * factor;
  }
}

// ---------------------------------------------------------------------------
// Kernel 4: per-head states gkv[d,e] = sum_s g(k)[s,d]*v[s,e] (and gmkv),
// plus column sums of g(k), g(-k). One block per (b,h); waves 0-3 -> gkv
// quadrants, waves 4-7 -> gmkv. L reduced in 32-chunks.
// ---------------------------------------------------------------------------
#define KSTRIDE 24
__global__ __launch_bounds__(256) void state_kernel(
    const float* __restrict__ K, const float* __restrict__ V,
    float* __restrict__ gkv, float* __restrict__ gmkv,
    float* __restrict__ gksum, float* __restrict__ gmksum)
{
  __shared__ bf16_t gkA [4 * 32 * KSTRIDE];  // A-type frags (m=d, k=s)
  __shared__ bf16_t gmkA[4 * 32 * KSTRIDE];
  __shared__ bf16_t vB  [4 * 32 * KSTRIDE];  // B-type frags (k=s, n=e)
  __shared__ float redk[4][64], redmk[4][64];

  int bh = blockIdx.x;
  int b = bh >> 4, h = bh & 15;
  int tid = threadIdx.x;
  int lane = tid & 31, w = tid >> 5;
  int sel = w >> 2;               // 0 -> gk path, 1 -> gmk path
  int wq = w & 3;
  int wm = wq & 1, wn = wq >> 1;  // 2x2 waves cover 64x64 output
  int mhalf = lane >> 4, l15 = lane & 15;

  v8f acc[2][2];
#pragma unroll
  for (int i = 0; i < 2; ++i)
#pragma unroll
    for (int j = 0; j < 2; ++j) zero_acc(acc[i][j]);
  float sgk = 0.f, sgmk = 0.f;  // this thread's d = tid&63 (fixed by mapping)

  for (int s0 = 0; s0 < LL; s0 += 32) {
#pragma unroll
    for (int j = 0; j < 4; ++j) {
      int idx = j * 256 + tid;
      int d = idx & 63, s = (idx >> 6) * 2;  // d constant per thread
      int gi = ((b * LL + s0 + s) * EE) + h * DD + d;
      float kv0 = K[gi], kv1 = K[gi + EE];
      float vv0 = V[gi], vv1 = V[gi + EE];
      float g0 = gfun(kv0), g1 = gfun(kv1);
      float q0 = gfun(-kv0), q1 = gfun(-kv1);
      sgk += g0 + g1;
      sgmk += q0 + q1;
      int lane_a = (d & 15) + (((s >> 3) & 1) << 4);
      int ea = (s & 7) | ((s >> 4) << 3);
      int fragd = d >> 4;
      store2(&gkA [(fragd * 32 + lane_a) * KSTRIDE + ea], g0, g1);
      store2(&gmkA[(fragd * 32 + lane_a) * KSTRIDE + ea], q0, q1);
      int lane_b = (d & 15) + (((s >> 3) & 1) << 4);
      store2(&vB[(fragd * 32 + lane_b) * KSTRIDE + ea], vv0, vv1);
    }
    __syncthreads();

    const bf16_t* Abase = sel ? gmkA : gkA;
#pragma unroll
    for (int ti = 0; ti < 2; ++ti) {
      v16bf afrag = load_frag(&Abase[((wm * 2 + ti) * 32 + lane) * KSTRIDE]);
#pragma unroll
      for (int tj = 0; tj < 2; ++tj) {
        v16bf bfrag = load_frag(&vB[((wn * 2 + tj) * 32 + lane) * KSTRIDE]);
        acc[ti][tj] = wmma_bf16(afrag, bfrag, acc[ti][tj]);
      }
    }
    __syncthreads();
  }

  float* outstate = sel ? gmkv : gkv;
#pragma unroll
  for (int ti = 0; ti < 2; ++ti)
#pragma unroll
    for (int tj = 0; tj < 2; ++tj) {
      int e = wn * 32 + tj * 16 + l15;
#pragma unroll
      for (int r = 0; r < 8; ++r) {
        int d = wm * 32 + ti * 16 + r + 8 * mhalf;
        outstate[bh * DD * DD + d * DD + e] = acc[ti][tj][r];
      }
    }
  // reduce the f32 column sums (4 partials per d)
  redk[tid >> 6][tid & 63] = sgk;
  redmk[tid >> 6][tid & 63] = sgmk;
  __syncthreads();
  if (tid < DD) {
    gksum[bh * DD + tid]  = redk[0][tid] + redk[1][tid] + redk[2][tid] + redk[3][tid];
    gmksum[bh * DD + tid] = redmk[0][tid] + redmk[1][tid] + redmk[2][tid] + redmk[3][tid];
  }
}

// ---------------------------------------------------------------------------
// Kernel 5: num = g(q) @ gkv + g(-q) @ gmkv ; den = g(q).gksum + g(-q).gmksum
// out = num / (den + 1e-6). Block tile 128(L) x 64(D) per (b,h).
// ---------------------------------------------------------------------------
#define QSTRIDE 16  // tighter stride to stay under 64KB static LDS
__global__ __launch_bounds__(256) void out_kernel(
    const float* __restrict__ Q,
    const float* __restrict__ gkv, const float* __restrict__ gmkv,
    const float* __restrict__ gksum, const float* __restrict__ gmksum,
    float* __restrict__ out)
{
  __shared__ bf16_t gqA [16 * 32 * QSTRIDE];  // 8 row-tiles x 2 ksteps (16 KB)
  __shared__ bf16_t gmqA[16 * 32 * QSTRIDE];  // 16 KB
  __shared__ bf16_t b0B [8 * 32 * QSTRIDE];   // 4 col-tiles x 2 ksteps (8 KB)
  __shared__ bf16_t b1B [8 * 32 * QSTRIDE];   // 8 KB
  __shared__ float denS[128];
  __shared__ float s0L[64], s1L[64];

  int bh = blockIdx.y;
  int b = bh >> 4, h = bh & 15;
  int l0 = blockIdx.x * 128;
  int tid = threadIdx.x;
  int lane = tid & 31, w = tid >> 5;
  int wm = w & 3, wn = w >> 2;
  int mhalf = lane >> 4, l15 = lane & 15;

  if (tid < 64) {
    s0L[tid] = gksum[bh * DD + tid];
    s1L[tid] = gmksum[bh * DD + tid];
  }
  if (tid < 128) denS[tid] = 0.f;
  __syncthreads();

  // Stage g(q), g(-q) into A-type frags; fold denominator partials in f32.
#pragma unroll
  for (int j = 0; j < 16; ++j) {
    int idx = j * 256 + tid;
    int row = idx >> 5, d = (idx & 31) * 2;
    float2 qv = *reinterpret_cast<const float2*>(
        &Q[((b * LL + l0 + row) * EE) + h * DD + d]);
    float g0 = gfun(qv.x), g1 = gfun(qv.y);
    float m0 = gfun(-qv.x), m1 = gfun(-qv.y);
    int kstep = d >> 5, c = d & 31;
    int lane_s = (row & 15) + (((c >> 3) & 1) << 4);
    int e = (c & 7) | ((c >> 4) << 3);
    int frag = (row >> 4) * 2 + kstep;
    store2(&gqA [(frag * 32 + lane_s) * QSTRIDE + e], g0, g1);
    store2(&gmqA[(frag * 32 + lane_s) * QSTRIDE + e], m0, m1);
    atomicAdd(&denS[row],
              g0 * s0L[d] + g1 * s0L[d + 1] + m0 * s1L[d] + m1 * s1L[d + 1]);
  }
  // Stage state matrices as B-type frags (k=d, n=e).
#pragma unroll
  for (int j = 0; j < 8; ++j) {
    int idx = j * 256 + tid;
    int n = idx & 63, d = (idx >> 6) * 2;
    int gi = bh * DD * DD + d * DD + n;
    float a0 = gkv[gi],  a1 = gkv[gi + DD];
    float c0 = gmkv[gi], c1 = gmkv[gi + DD];
    int lane_s = (n & 15) + (((d >> 3) & 1) << 4);
    int frag = (n >> 4) * 2 + (d >> 5);
    store2(&b0B[(frag * 32 + lane_s) * QSTRIDE + ((d & 7) | (((d >> 4) & 1) << 3))], a0, a1);
    store2(&b1B[(frag * 32 + lane_s) * QSTRIDE + ((d & 7) | (((d >> 4) & 1) << 3))], c0, c1);
  }
  __syncthreads();

  v8f acc[2][2];
#pragma unroll
  for (int i = 0; i < 2; ++i)
#pragma unroll
    for (int j = 0; j < 2; ++j) zero_acc(acc[i][j]);

#pragma unroll
  for (int kstep = 0; kstep < 2; ++kstep) {
#pragma unroll
    for (int ti = 0; ti < 2; ++ti) {
      int afrag_id = (wm * 2 + ti) * 2 + kstep;
      v16bf aq  = load_frag(&gqA [(afrag_id * 32 + lane) * QSTRIDE]);
      v16bf amq = load_frag(&gmqA[(afrag_id * 32 + lane) * QSTRIDE]);
#pragma unroll
      for (int tj = 0; tj < 2; ++tj) {
        int bfrag_id = (wn * 2 + tj) * 2 + kstep;
        v16bf b0f = load_frag(&b0B[(bfrag_id * 32 + lane) * QSTRIDE]);
        v16bf b1f = load_frag(&b1B[(bfrag_id * 32 + lane) * QSTRIDE]);
        acc[ti][tj] = wmma_bf16(aq,  b0f, acc[ti][tj]);
        acc[ti][tj] = wmma_bf16(amq, b1f, acc[ti][tj]);
      }
    }
  }

#pragma unroll
  for (int ti = 0; ti < 2; ++ti)
#pragma unroll
    for (int tj = 0; tj < 2; ++tj) {
      int e = wn * 32 + tj * 16 + l15;
#pragma unroll
      for (int r = 0; r < 8; ++r) {
        int row = wm * 32 + ti * 16 + r + 8 * mhalf;
        float den = denS[row] + 1e-6f;
        out[((b * LL + l0 + row) * EE) + h * DD + e] = acc[ti][tj][r] / den;
      }
    }
}

// ---------------------------------------------------------------------------
extern "C" void kernel_launch(void* const* d_in, const int* in_sizes, int n_in,
                              void* d_out, int out_size, void* d_ws, size_t ws_size,
                              hipStream_t stream) {
  const float* x  = (const float*)d_in[0];
  const float* Wq = (const float*)d_in[1];
  const float* bq = (const float*)d_in[2];
  const float* Wk = (const float*)d_in[3];
  const float* bk = (const float*)d_in[4];
  const float* Wv = (const float*)d_in[5];
  const float* bv = (const float*)d_in[6];
  float* out = (float*)d_out;
  float* ws = (float*)d_ws;

  // Workspace (floats): Q,K,V [B*L*E each] + qg + states + sums (~204 MB)
  const size_t QKV = (size_t)BB * LL * EE;
  float* Q      = ws;
  float* K      = ws + QKV;
  float* V      = ws + 2 * QKV;
  float* qg     = ws + 3 * QKV;
  float* gkv    = qg + (size_t)BB * HH * DD;
  float* gmkv   = gkv + (size_t)BB * HH * DD * DD;
  float* gksum  = gmkv + (size_t)BB * HH * DD * DD;
  float* gmksum = gksum + (size_t)BB * HH * DD;

  dim3 g1(EE / 64, (BB * LL) / 128, 3);
  qkv_gemm<<<g1, 256, 0, stream>>>(x, Wq, bq, Wk, bk, Wv, bv, Q, K, V);
  qglobal_kernel<<<BB * HH, 256, 0, stream>>>(Q, qg);
  softmax_scale_k<<<BB * HH, 256, 0, stream>>>(qg, K);
  state_kernel<<<BB * HH, 256, 0, stream>>>(K, V, gkv, gmkv, gksum, gmksum);
  dim3 g5(LL / 128, BB * HH);
  out_kernel<<<g5, 256, 0, stream>>>(Q, gkv, gmkv, gksum, gmksum, out);
}
